// GCN3D_11948599018347
// MI455X (gfx1250) — matrix operand order repaired and checked by reading, not compiled, not run on hardware
//
#include <hip/hip_runtime.h>
#include <math.h>

// ---------------------------------------------------------------------------
// MI455X (gfx1250) implementation of the GCN3D reference network.
// All dense contractions go through v_wmma_f32_16x16x32_f16 (f16 in, f32 acc),
// 32x32 output tile per wave (4 WMMAs / k-step), float4 fast-path loads.
// Irregular work (KNN top-k, gathers, softmax, stats, MT19937 perm) is scalar.
// ---------------------------------------------------------------------------

typedef __attribute__((ext_vector_type(16))) _Float16 v16h;
typedef __attribute__((ext_vector_type(8)))  float    v8f;

#define FLT_BIG 3.402823466e38f

// ------------------------------ WMMA GEMM ----------------------------------
// C[M,N] = op(A)[M,K] * op(B)[K,N] + bias[N]
// TA: element (m,k) = A[k*lda + m]   (else A[m*lda + k])
// TB: element (k,n) = B[n*ldb + k]   (else B[k*ldb + n])

// A fragment: 16x32 f16: lanes0-15 K=0..15, lanes16-31 K=16..31
template <int TRANS>
__device__ __forceinline__ void loadAfrag(v16h& a, const float* __restrict__ A,
                                          int am, int k0, int hl, int M, int K, int lda) {
  if ((k0 + 32 <= K) && (am < M)) {
    if (!TRANS) {
      const float* pa = A + (size_t)am * lda + (k0 + hl * 16);
      const float4 f0 = *reinterpret_cast<const float4*>(pa);
      const float4 f1 = *reinterpret_cast<const float4*>(pa + 4);
      const float4 f2 = *reinterpret_cast<const float4*>(pa + 8);
      const float4 f3 = *reinterpret_cast<const float4*>(pa + 12);
      a[0] = (_Float16)f0.x;  a[1] = (_Float16)f0.y;  a[2]  = (_Float16)f0.z;  a[3]  = (_Float16)f0.w;
      a[4] = (_Float16)f1.x;  a[5] = (_Float16)f1.y;  a[6]  = (_Float16)f1.z;  a[7]  = (_Float16)f1.w;
      a[8] = (_Float16)f2.x;  a[9] = (_Float16)f2.y;  a[10] = (_Float16)f2.z;  a[11] = (_Float16)f2.w;
      a[12] = (_Float16)f3.x; a[13] = (_Float16)f3.y; a[14] = (_Float16)f3.z;  a[15] = (_Float16)f3.w;
    } else {
#pragma unroll
      for (int i = 0; i < 8; ++i) {
        a[i]     = (_Float16)A[(size_t)(k0 + hl * 16 + i) * lda + am];
        a[i + 8] = (_Float16)A[(size_t)(k0 + hl * 16 + 8 + i) * lda + am];
      }
    }
  } else {
#pragma unroll
    for (int i = 0; i < 8; ++i) {
      int k1 = k0 + hl * 16 + i, k2 = k0 + hl * 16 + 8 + i;
      a[i]     = (am < M && k1 < K)
                   ? (_Float16)(TRANS ? A[(size_t)k1 * lda + am] : A[(size_t)am * lda + k1])
                   : (_Float16)0.f;
      a[i + 8] = (am < M && k2 < K)
                   ? (_Float16)(TRANS ? A[(size_t)k2 * lda + am] : A[(size_t)am * lda + k2])
                   : (_Float16)0.f;
    }
  }
}

// B fragment: 32x16 f16: lanes0-15 K=0..15, lanes16-31 K=16..31 (col = lane&15)
template <int TRANS>
__device__ __forceinline__ void loadBfrag(v16h& b, const float* __restrict__ Bm,
                                          int bn, int k0, int hl, int N, int K, int ldb) {
  if ((k0 + 32 <= K) && (bn < N)) {
    if (TRANS) {
      const float* pb = Bm + (size_t)bn * ldb + (k0 + hl * 16);
      const float4 f0 = *reinterpret_cast<const float4*>(pb);
      const float4 f1 = *reinterpret_cast<const float4*>(pb + 4);
      const float4 f2 = *reinterpret_cast<const float4*>(pb + 8);
      const float4 f3 = *reinterpret_cast<const float4*>(pb + 12);
      b[0] = (_Float16)f0.x;  b[1] = (_Float16)f0.y;  b[2]  = (_Float16)f0.z;  b[3]  = (_Float16)f0.w;
      b[4] = (_Float16)f1.x;  b[5] = (_Float16)f1.y;  b[6]  = (_Float16)f1.z;  b[7]  = (_Float16)f1.w;
      b[8] = (_Float16)f2.x;  b[9] = (_Float16)f2.y;  b[10] = (_Float16)f2.z;  b[11] = (_Float16)f2.w;
      b[12] = (_Float16)f3.x; b[13] = (_Float16)f3.y; b[14] = (_Float16)f3.z;  b[15] = (_Float16)f3.w;
    } else {
#pragma unroll
      for (int i = 0; i < 16; ++i)
        b[i] = (_Float16)Bm[(size_t)(k0 + hl * 16 + i) * ldb + bn];
    }
  } else {
#pragma unroll
    for (int i = 0; i < 16; ++i) {
      int kk = k0 + hl * 16 + i;
      b[i] = (bn < N && kk < K)
               ? (_Float16)(TRANS ? Bm[(size_t)bn * ldb + kk] : Bm[(size_t)kk * ldb + bn])
               : (_Float16)0.f;
    }
  }
}

// C tile store: row = row0 + hl*8 + r, col = col0 + lr (ISA 7.12.2 C/D layout)
__device__ __forceinline__ void storeCtile(const v8f& acc, float* __restrict__ C,
                                           const float* __restrict__ bias,
                                           int row0, int col0, int hl, int lr,
                                           int M, int N, int ldc) {
  int col = col0 + lr;
  float bv = (bias != nullptr && col < N) ? bias[col] : 0.f;
  if (row0 + 16 <= M && col < N) {
#pragma unroll
    for (int r = 0; r < 8; ++r)
      C[(size_t)(row0 + hl * 8 + r) * ldc + col] = acc[r] + bv;
  } else {
#pragma unroll
    for (int r = 0; r < 8; ++r) {
      int row = row0 + hl * 8 + r;
      if (row < M && col < N) C[(size_t)row * ldc + col] = acc[r] + bv;
    }
  }
}

template <int TA, int TB>
__global__ void wmma_gemm_kernel(const float* __restrict__ A, const float* __restrict__ Bm,
                                 const float* __restrict__ bias, float* __restrict__ C,
                                 int M, int N, int K, int lda, int ldb, int ldc) {
  int lane = (int)(threadIdx.x & 31u);
  int hl = lane >> 4;
  int lr = lane & 15;
  int row0 = blockIdx.y * 32;
  int col0 = blockIdx.x * 32;
  int am0 = row0 + lr, am1 = row0 + 16 + lr;
  int bn0 = col0 + lr, bn1 = col0 + 16 + lr;
  v8f acc00 = {0.f, 0.f, 0.f, 0.f, 0.f, 0.f, 0.f, 0.f};
  v8f acc01 = acc00, acc10 = acc00, acc11 = acc00;
  for (int k0 = 0; k0 < K; k0 += 32) {
    v16h a0, a1, b0, b1;
    loadAfrag<TA>(a0, A, am0, k0, hl, M, K, lda);
    loadAfrag<TA>(a1, A, am1, k0, hl, M, K, lda);
    loadBfrag<TB>(b0, Bm, bn0, k0, hl, N, K, ldb);
    loadBfrag<TB>(b1, Bm, bn1, k0, hl, N, K, ldb);
    acc00 = __builtin_amdgcn_wmma_f32_16x16x32_f16(false, a0, false, b0, (short)0, acc00, false, false);
    acc01 = __builtin_amdgcn_wmma_f32_16x16x32_f16(false, a0, false, b1, (short)0, acc01, false, false);
    acc10 = __builtin_amdgcn_wmma_f32_16x16x32_f16(false, a1, false, b0, (short)0, acc10, false, false);
    acc11 = __builtin_amdgcn_wmma_f32_16x16x32_f16(false, a1, false, b1, (short)0, acc11, false, false);
  }
  storeCtile(acc00, C, bias, row0,      col0,      hl, lr, M, N, ldc);
  storeCtile(acc01, C, bias, row0,      col0 + 16, hl, lr, M, N, ldc);
  storeCtile(acc10, C, bias, row0 + 16, col0,      hl, lr, M, N, ldc);
  storeCtile(acc11, C, bias, row0 + 16, col0 + 16, hl, lr, M, N, ldc);
}

// --------------------------- vertex transpose ------------------------------
__global__ void transpose_verts_kernel(const float* __restrict__ in, float* __restrict__ out,
                                       int B, int V) {
  int t = blockIdx.x * blockDim.x + threadIdx.x;
  if (t >= B * V * 3) return;
  int b = t / (V * 3);
  int rem = t - b * V * 3;
  int i = rem / 3, d = rem - i * 3;
  out[t] = in[((size_t)b * 3 + d) * V + i];
}

// --------------------------------- KNN -------------------------------------
// One thread per query point, stable (dist,index) insertion -> matches top_k ties.
__global__ void knn_kernel(const float* __restrict__ v, int* __restrict__ out,
                           int B, int V, int K) {
  int t = blockIdx.x * blockDim.x + threadIdx.x;
  if (t >= B * V) return;
  int b = t / V, i = t - b * V;
  const float* vb = v + (size_t)b * V * 3;
  float xi = vb[i * 3 + 0], yi = vb[i * 3 + 1], zi = vb[i * 3 + 2];
  float sqi = xi * xi + yi * yi + zi * zi;
  float bd[21];
  int   bi[21];
  int cnt = 0;
  const int kp1 = K + 1;   // <= 21
  for (int j = 0; j < V; ++j) {
    float xj = vb[j * 3 + 0], yj = vb[j * 3 + 1], zj = vb[j * 3 + 2];
    float inner = xi * xj + yi * yj + zi * zj;
    float sqj = xj * xj + yj * yj + zj * zj;
    float d = sqi - 2.f * inner + sqj;
    if (cnt < kp1) {
      int pos = cnt;
      while (pos > 0 && bd[pos - 1] > d) { bd[pos] = bd[pos - 1]; bi[pos] = bi[pos - 1]; --pos; }
      bd[pos] = d; bi[pos] = j; ++cnt;
    } else if (d < bd[kp1 - 1]) {
      int pos = kp1 - 1;
      while (pos > 0 && bd[pos - 1] > d) { bd[pos] = bd[pos - 1]; bi[pos] = bi[pos - 1]; --pos; }
      bd[pos] = d; bi[pos] = j;
    }
  }
  int* o = out + ((size_t)b * V + i) * K;
  for (int n = 0; n < K; ++n) o[n] = bi[n + 1];   // drop self (first)
}

// ---------------------------- conv_surface ---------------------------------
__global__ void conv_surface_kernel(const int* __restrict__ idx, const float* __restrict__ v,
                                    const float* __restrict__ dirs, float* __restrict__ fm0,
                                    int B, int V, int K, int M) {
  int p = blockIdx.x;
  int b = p / V, i = p - b * V;
  __shared__ float dn[20 * 3];
  const float* vb = v + (size_t)b * V * 3;
  float xi = vb[i * 3 + 0], yi = vb[i * 3 + 1], zi = vb[i * 3 + 2];
  const int* id = idx + ((size_t)b * V + i) * K;
  if ((int)threadIdx.x < K) {
    int n = threadIdx.x;
    int j = id[n];
    float dx = vb[j * 3 + 0] - xi, dy = vb[j * 3 + 1] - yi, dz = vb[j * 3 + 2] - zi;
    float nr = fmaxf(sqrtf(dx * dx + dy * dy + dz * dz), 1e-12f);
    dn[n * 3 + 0] = dx / nr; dn[n * 3 + 1] = dy / nr; dn[n * 3 + 2] = dz / nr;
  }
  __syncthreads();
  int m = threadIdx.x;
  if (m < M) {
    float d0 = dirs[m], d1 = dirs[M + m], d2 = dirs[2 * M + m];
    float nn = fmaxf(sqrtf(d0 * d0 + d1 * d1 + d2 * d2), 1e-12f);
    d0 /= nn; d1 /= nn; d2 /= nn;
    float best = -FLT_BIG;
    for (int n = 0; n < K; ++n) {
      float th = fmaxf(0.f, dn[n * 3] * d0 + dn[n * 3 + 1] * d1 + dn[n * 3 + 2] * d2);
      best = fmaxf(best, th);
    }
    fm0[((size_t)b * V + i) * M + m] = fmaxf(best, 0.f);
  }
}

// -------------------------- conv_layer combine -----------------------------
__global__ void conv_combine_kernel(const int* __restrict__ idx, const float* __restrict__ v,
                                    const float* __restrict__ dirs, const float* __restrict__ fo,
                                    float* __restrict__ out, int B, int V, int K, int C) {
  int p = blockIdx.x;
  int b = p / V, i = p - b * V;
  __shared__ float dn[20 * 3];
  __shared__ int   nid[20];
  const float* vb = v + (size_t)b * V * 3;
  float xi = vb[i * 3 + 0], yi = vb[i * 3 + 1], zi = vb[i * 3 + 2];
  const int* id = idx + ((size_t)b * V + i) * K;
  if ((int)threadIdx.x < K) {
    int n = threadIdx.x;
    int j = id[n];
    nid[n] = j;
    float dx = vb[j * 3 + 0] - xi, dy = vb[j * 3 + 1] - yi, dz = vb[j * 3 + 2] - zi;
    float nr = fmaxf(sqrtf(dx * dx + dy * dy + dz * dz), 1e-12f);
    dn[n * 3 + 0] = dx / nr; dn[n * 3 + 1] = dy / nr; dn[n * 3 + 2] = dz / nr;
  }
  __syncthreads();
  size_t rowBase = ((size_t)b * V + i) * (size_t)(2 * C);
  for (int c = threadIdx.x; c < C; c += blockDim.x) {
    float d0 = dirs[c], d1 = dirs[C + c], d2 = dirs[2 * C + c];
    float nn = fmaxf(sqrtf(d0 * d0 + d1 * d1 + d2 * d2), 1e-12f);
    d0 /= nn; d1 /= nn; d2 /= nn;
    float fc = fo[rowBase + c];
    float best = -FLT_BIG;
    for (int n = 0; n < K; ++n) {
      float th = fmaxf(0.f, dn[n * 3] * d0 + dn[n * 3 + 1] * d1 + dn[n * 3 + 2] * d2);
      float fs = fo[((size_t)b * V + nid[n]) * (size_t)(2 * C) + C + c];
      best = fmaxf(best, th * fs);
    }
    out[((size_t)b * V + i) * C + c] = fc + best;
  }
}

// --------------------- per-channel stats (mean / 1/std) --------------------
__global__ void colstats_kernel(const float* __restrict__ x, float* __restrict__ meanOut,
                                float* __restrict__ istdOut, int R, int C, float eps) {
  __shared__ float ssum[256];
  __shared__ float ssq[256];
  int c = blockIdx.x;
  int tid = threadIdx.x;
  float s = 0.f, q = 0.f;
  for (int r = tid; r < R; r += 256) { float vv = x[(size_t)r * C + c]; s += vv; q += vv * vv; }
  ssum[tid] = s; ssq[tid] = q;
  __syncthreads();
  for (int st = 128; st > 0; st >>= 1) {
    if (tid < st) { ssum[tid] += ssum[tid + st]; ssq[tid] += ssq[tid + st]; }
    __syncthreads();
  }
  if (tid == 0) {
    float m = ssum[0] / (float)R;
    float var = fmaxf(ssq[0] / (float)R - m * m, 0.f);
    meanOut[c] = m;
    istdOut[c] = rsqrtf(var + eps);
  }
}

// out = [addBase +] relu?( gamma*(x-mean)*istd + beta )
__global__ void norm_apply_kernel(const float* __restrict__ x, const float* __restrict__ mean,
                                  const float* __restrict__ istd, const float* __restrict__ gamma,
                                  const float* __restrict__ beta, const float* __restrict__ addBase,
                                  float* __restrict__ out, size_t total, int C, int relu) {
  size_t t = (size_t)blockIdx.x * blockDim.x + threadIdx.x;
  if (t >= total) return;
  int c = (int)(t % (size_t)C);
  float val = (x[t] - mean[c]) * istd[c];
  if (gamma != nullptr) val = val * gamma[c] + beta[c];
  if (relu) val = fmaxf(val, 0.f);
  if (addBase != nullptr) val += addBase[t];
  out[t] = val;
}

// ------------------------------ elementwise --------------------------------
__global__ void ew_add_relu_kernel(float* __restrict__ o, const float* __restrict__ a,
                                   const float* __restrict__ b, size_t n) {
  size_t t = (size_t)blockIdx.x * blockDim.x + threadIdx.x;
  if (t < n) o[t] = fmaxf(a[t] + b[t], 0.f);
}
__global__ void ew_add_kernel(float* __restrict__ o, const float* __restrict__ a,
                              const float* __restrict__ b, size_t n) {
  size_t t = (size_t)blockIdx.x * blockDim.x + threadIdx.x;
  if (t < n) o[t] = a[t] + b[t];
}
__global__ void ew_sub_kernel(float* __restrict__ o, const float* __restrict__ a,
                              const float* __restrict__ b, size_t n) {
  size_t t = (size_t)blockIdx.x * blockDim.x + threadIdx.x;
  if (t < n) o[t] = a[t] - b[t];
}

// ------------------------------- attention ---------------------------------
__global__ void softmax_rows_kernel(float* __restrict__ e, int V) {
  __shared__ float red[256];
  int n = blockIdx.x, tid = threadIdx.x;
  float* row = e + (size_t)n * V;
  float mx = -FLT_BIG;
  for (int m = tid; m < V; m += 256) mx = fmaxf(mx, row[m]);
  red[tid] = mx; __syncthreads();
  for (int st = 128; st > 0; st >>= 1) { if (tid < st) red[tid] = fmaxf(red[tid], red[tid + st]); __syncthreads(); }
  mx = red[0]; __syncthreads();
  float sum = 0.f;
  for (int m = tid; m < V; m += 256) { float ex = expf(row[m] - mx); row[m] = ex; sum += ex; }
  red[tid] = sum; __syncthreads();
  for (int st = 128; st > 0; st >>= 1) { if (tid < st) red[tid] += red[tid + st]; __syncthreads(); }
  float inv = 1.f / red[0];
  for (int m = tid; m < V; m += 256) row[m] *= inv;
}

__global__ void colsum_kernel(const float* __restrict__ e, float* __restrict__ cs, int V) {
  int m = blockIdx.x * blockDim.x + threadIdx.x;
  if (m >= V) return;
  float s = 0.f;
  for (int n = 0; n < V; ++n) s += e[(size_t)n * V + m];
  cs[m] = s;
}

// xr[m, o] /= (1e-9 + colsum[m])
__global__ void scale_rows_kernel(float* __restrict__ x, const float* __restrict__ cs,
                                  int V, int C) {
  size_t t = (size_t)blockIdx.x * blockDim.x + threadIdx.x;
  if (t >= (size_t)V * C) return;
  int m = (int)(t / (size_t)C);
  x[t] /= (1e-9f + cs[m]);
}

// ------------------------------- pooling -----------------------------------
__global__ void pool_kernel(const float* __restrict__ fm, const float* __restrict__ v,
                            const int* __restrict__ idxp, const int* __restrict__ perm,
                            float* __restrict__ fmOut, float* __restrict__ vOut,
                            int B, int Vin, int Vout, int C) {
  int p = blockIdx.x;
  int b = p / Vout, r = p - b * Vout;
  int sIdx = perm[r];
  if ((int)threadIdx.x < 3)
    vOut[((size_t)b * Vout + r) * 3 + threadIdx.x] = v[((size_t)b * Vin + sIdx) * 3 + threadIdx.x];
  const int* id = idxp + ((size_t)b * Vin + sIdx) * 4;
  int j0 = id[0], j1 = id[1], j2 = id[2], j3 = id[3];
  for (int c = threadIdx.x; c < C; c += blockDim.x) {
    float m0 = fm[((size_t)b * Vin + j0) * C + c];
    m0 = fmaxf(m0, fm[((size_t)b * Vin + j1) * C + c]);
    m0 = fmaxf(m0, fm[((size_t)b * Vin + j2) * C + c]);
    m0 = fmaxf(m0, fm[((size_t)b * Vin + j3) * C + c]);
    fmOut[((size_t)b * Vout + r) * C + c] = m0;
  }
}

__global__ void maxover_points_kernel(const float* __restrict__ fm, float* __restrict__ g,
                                      int B, int V, int C) {
  int t = blockIdx.x * blockDim.x + threadIdx.x;
  if (t >= B * C) return;
  int b = t / C, c = t - b * C;
  float mx = -FLT_BIG;
  for (int i = 0; i < V; ++i) mx = fmaxf(mx, fm[((size_t)b * V + i) * C + c]);
  g[t] = mx;
}

// -------------------- exact numpy RandomState permutation -------------------
struct MTState { unsigned mt[624]; int mti; };

__device__ unsigned mt_next(MTState* st) {
  if (st->mti >= 624) {
    for (int i = 0; i < 624; ++i) {
      unsigned y = (st->mt[i] & 0x80000000u) | (st->mt[(i + 1) % 624] & 0x7fffffffu);
      st->mt[i] = st->mt[(i + 397) % 624] ^ (y >> 1) ^ ((y & 1u) ? 0x9908b0dfu : 0u);
    }
    st->mti = 0;
  }
  unsigned y = st->mt[st->mti++];
  y ^= y >> 11;
  y ^= (y << 7) & 0x9d2c5680u;
  y ^= (y << 15) & 0xefc60000u;
  y ^= y >> 18;
  return y;
}

__global__ void perm_kernel(int* __restrict__ out, int n, unsigned seed) {
  if (blockIdx.x != 0 || threadIdx.x != 0) return;
  MTState st;
  unsigned s = seed;
  for (int pos = 0; pos < 624; ++pos) {
    st.mt[pos] = s;
    s = 1812433253u * (s ^ (s >> 30)) + (unsigned)(pos + 1);
  }
  st.mti = 624;
  for (int i = 0; i < n; ++i) out[i] = i;
  for (int i = n - 1; i > 0; --i) {    // Fisher-Yates, numpy random_interval
    unsigned mask = (unsigned)i;
    mask |= mask >> 1; mask |= mask >> 2; mask |= mask >> 4; mask |= mask >> 8; mask |= mask >> 16;
    unsigned j;
    do { j = mt_next(&st) & mask; } while (j > (unsigned)i);
    int tmp = out[i]; out[i] = out[j]; out[j] = tmp;
  }
}

// ------------------------------ host helpers -------------------------------
static inline unsigned cdivu(size_t a, unsigned b) { return (unsigned)((a + b - 1) / b); }

static void gemm(hipStream_t s, const float* A, const float* B, const float* bias, float* C,
                 int M, int N, int K, int lda, int ldb, int ldc, int tA, int tB) {
  dim3 grid(cdivu((size_t)N, 32), cdivu((size_t)M, 32));
  if (!tA && !tB)
    wmma_gemm_kernel<0, 0><<<grid, dim3(32), 0, s>>>(A, B, bias, C, M, N, K, lda, ldb, ldc);
  else if (!tA && tB)
    wmma_gemm_kernel<0, 1><<<grid, dim3(32), 0, s>>>(A, B, bias, C, M, N, K, lda, ldb, ldc);
  else
    wmma_gemm_kernel<1, 0><<<grid, dim3(32), 0, s>>>(A, B, bias, C, M, N, K, lda, ldb, ldc);
}

static float* allocF(char* base, size_t& off, size_t n) {
  float* p = (float*)(base + off);
  off += (n * sizeof(float) + 255) & ~(size_t)255;
  return p;
}
static int* allocI(char* base, size_t& off, size_t n) {
  int* p = (int*)(base + off);
  off += (n * sizeof(int) + 255) & ~(size_t)255;
  return p;
}

// graph conv block: fo = fmIn@w + b ; combine ; bn ; relu -> fmOut
static void run_conv(hipStream_t s, const int* idx, const float* v, const float* fmIn,
                     const float* w, const float* bias, const float* dirs,
                     int B, int V, int K, int cin, int cout,
                     float* fo, float* convA, float* meanb, float* istdb, float* fmOut) {
  int R = B * V;
  gemm(s, fmIn, w, bias, fo, R, 2 * cout, cin, cin, 2 * cout, 2 * cout, 0, 0);
  conv_combine_kernel<<<dim3(R), dim3(128), 0, s>>>(idx, v, dirs, fo, convA, B, V, K, cout);
  colstats_kernel<<<dim3(cout), dim3(256), 0, s>>>(convA, meanb, istdb, R, cout, 1e-5f);
  size_t tot = (size_t)R * cout;
  norm_apply_kernel<<<dim3(cdivu(tot, 256)), dim3(256), 0, s>>>(
      convA, meanb, istdb, nullptr, nullptr, nullptr, fmOut, tot, cout, 1);
}

// self-attention block (in-place on fm): fm = fm + relu(norm(wt@(x-xr)+bt))
static void run_sa(hipStream_t s, float* fm, int B, int V, int C,
                   const float* wq, const float* wv, const float* bv,
                   const float* wt, const float* bt, const float* gg, const float* bb,
                   float* xq, float* xvt, float* xrt, float* diffb, float* tt,
                   float* energy, float* cs, float* meanb, float* istdb) {
  int R = B * V, Q = C / 4;
  gemm(s, fm, wq, nullptr, xq, R, Q, C, C, C, Q, 0, 1);        // xq = fm @ wq^T
  gemm(s, fm, wv, bv, xvt, R, C, C, C, C, C, 0, 1);            // xv^T = fm @ wv^T + bv
  for (int b = 0; b < B; ++b) {
    const float* xqb = xq + (size_t)b * V * Q;
    gemm(s, xqb, xqb, nullptr, energy, V, V, Q, Q, Q, V, 0, 1); // energy = Xq Xq^T
    softmax_rows_kernel<<<dim3(V), dim3(256), 0, s>>>(energy, V);
    colsum_kernel<<<dim3(cdivu((size_t)V, 256)), dim3(256), 0, s>>>(energy, cs, V);
    float* xrb = xrt + (size_t)b * V * C;
    gemm(s, energy, xvt + (size_t)b * V * C, nullptr, xrb, V, C, V, V, C, C, 1, 0); // attn^T @ xv^T
    scale_rows_kernel<<<dim3(cdivu((size_t)V * C, 256)), dim3(256), 0, s>>>(xrb, cs, V, C);
  }
  size_t tot = (size_t)R * C;
  ew_sub_kernel<<<dim3(cdivu(tot, 256)), dim3(256), 0, s>>>(diffb, fm, xrt, tot);
  gemm(s, diffb, wt, bt, tt, R, C, C, C, C, C, 0, 1);          // t^T = diff @ wt^T + bt
  colstats_kernel<<<dim3(C), dim3(256), 0, s>>>(tt, meanb, istdb, R, C, 1e-5f);
  norm_apply_kernel<<<dim3(cdivu(tot, 256)), dim3(256), 0, s>>>(
      tt, meanb, istdb, gg, bb, fm, fm, tot, C, 1);            // fm += relu(norm(t))
}

// ------------------------------- entry point -------------------------------
extern "C" void kernel_launch(void* const* d_in, const int* in_sizes, int n_in,
                              void* d_out, int out_size, void* d_ws, size_t ws_size,
                              hipStream_t stream) {
  (void)in_sizes; (void)n_in; (void)out_size; (void)ws_size;
  const int B = 8, V1 = 2048, V2 = 512, V3 = 128, KNN = 20;

  const float* vertices = (const float*)d_in[0];
  const float* dir0 = (const float*)d_in[1];
  const float* w1 = (const float*)d_in[2];  const float* b1 = (const float*)d_in[3];  const float* dir1 = (const float*)d_in[4];
  const float* w2 = (const float*)d_in[5];  const float* b2 = (const float*)d_in[6];  const float* dir2 = (const float*)d_in[7];
  const float* w3 = (const float*)d_in[8];  const float* b3 = (const float*)d_in[9];  const float* dir3 = (const float*)d_in[10];
  const float* w4 = (const float*)d_in[11]; const float* b4 = (const float*)d_in[12]; const float* dir4 = (const float*)d_in[13];
  const float* d1 = (const float*)d_in[14]; const float* d2 = (const float*)d_in[15];
  const float* d3 = (const float*)d_in[16]; const float* d4 = (const float*)d_in[17];
  const float* sa[4][7];
  for (int si = 0; si < 4; ++si)
    for (int k = 0; k < 7; ++k) sa[si][k] = (const float*)d_in[18 + si * 7 + k];
  const float* cw1 = (const float*)d_in[46]; const float* cb1 = (const float*)d_in[47];
  const float* cg  = (const float*)d_in[48]; const float* cbeta = (const float*)d_in[49];
  const float* cw2 = (const float*)d_in[50]; const float* cb2 = (const float*)d_in[51];
  float* outp = (float*)d_out;

  // ---- workspace layout (~85 MB) ----
  char* base = (char*)d_ws; size_t off = 0;
  float* v1   = allocF(base, off, (size_t)B * V1 * 3);
  float* v2b  = allocF(base, off, (size_t)B * V2 * 3);
  float* v3b  = allocF(base, off, (size_t)B * V3 * 3);
  int*   idx1 = allocI(base, off, (size_t)B * V1 * KNN);
  int*   idxp1= allocI(base, off, (size_t)B * V1 * 4);
  int*   idx2 = allocI(base, off, (size_t)B * V2 * KNN);
  int*   idxp2= allocI(base, off, (size_t)B * V2 * 4);
  int*   idx3 = allocI(base, off, (size_t)B * V3 * KNN);
  int*   perm1= allocI(base, off, V1);
  int*   perm2= allocI(base, off, V2);
  float* fm0  = allocF(base, off, (size_t)B * V1 * 32);
  float* res1 = allocF(base, off, (size_t)B * V1 * 64);
  float* fm1  = allocF(base, off, (size_t)B * V1 * 64);
  float* fm1p = allocF(base, off, (size_t)B * V2 * 64);
  float* fm2  = allocF(base, off, (size_t)B * V2 * 128);
  float* res2 = allocF(base, off, (size_t)B * V2 * 128);
  float* fm3  = allocF(base, off, (size_t)B * V2 * 256);
  float* res3 = allocF(base, off, (size_t)B * V2 * 256);
  float* fm3p = allocF(base, off, (size_t)B * V3 * 256);
  float* fm4  = allocF(base, off, (size_t)B * V3 * 1024);
  float* res4 = allocF(base, off, (size_t)B * V3 * 1024);
  float* fo   = allocF(base, off, (size_t)B * V1 * 128);   // max fo across stages (2M)
  float* convA= allocF(base, off, (size_t)B * V1 * 64);    // max conv out (1M)
  float* xq   = allocF(base, off, 262144);
  float* xvt  = allocF(base, off, 1048576);
  float* xrt  = allocF(base, off, 1048576);
  float* diffb= allocF(base, off, 1048576);
  float* tt   = allocF(base, off, 1048576);
  float* energy = allocF(base, off, (size_t)V1 * V1);      // 16 MB, per-batch reuse
  float* cs   = allocF(base, off, V1);
  float* meanb= allocF(base, off, 2048);
  float* istdb= allocF(base, off, 2048);
  float* gbuf = allocF(base, off, (size_t)B * 1024);
  float* hbuf = allocF(base, off, (size_t)B * 256);

  // ---- stage 0: layout + graph + surface features ----
  transpose_verts_kernel<<<dim3(cdivu((size_t)B * V1 * 3, 256)), dim3(256), 0, stream>>>(vertices, v1, B, V1);
  perm_kernel<<<dim3(1), dim3(1), 0, stream>>>(perm1, V1, 0u);
  perm_kernel<<<dim3(1), dim3(1), 0, stream>>>(perm2, V2, 1u);
  knn_kernel<<<dim3(cdivu((size_t)B * V1, 128)), dim3(128), 0, stream>>>(v1, idx1, B, V1, KNN);
  conv_surface_kernel<<<dim3(B * V1), dim3(32), 0, stream>>>(idx1, v1, dir0, fm0, B, V1, KNN, 32);

  // ---- stage 1 (2048 pts, 32->64) ----
  gemm(stream, fm0, d1, nullptr, res1, B * V1, 64, 32, 32, 64, 64, 0, 0);
  run_conv(stream, idx1, v1, fm0, w1, b1, dir1, B, V1, KNN, 32, 64, fo, convA, meanb, istdb, fm1);
  run_sa(stream, fm1, B, V1, 64, sa[0][0], sa[0][1], sa[0][2], sa[0][3], sa[0][4], sa[0][5], sa[0][6],
         xq, xvt, xrt, diffb, tt, energy, cs, meanb, istdb);
  ew_add_relu_kernel<<<dim3(cdivu((size_t)B * V1 * 64, 256)), dim3(256), 0, stream>>>(fm1, fm1, res1, (size_t)B * V1 * 64);

  // ---- pool 1: 2048 -> 512 ----
  knn_kernel<<<dim3(cdivu((size_t)B * V1, 128)), dim3(128), 0, stream>>>(v1, idxp1, B, V1, 4);
  pool_kernel<<<dim3(B * V2), dim3(128), 0, stream>>>(fm1, v1, idxp1, perm1, fm1p, v2b, B, V1, V2, 64);
  knn_kernel<<<dim3(cdivu((size_t)B * V2, 128)), dim3(128), 0, stream>>>(v2b, idx2, B, V2, KNN);

  // ---- stage 2 (512 pts, 64->128) ----
  gemm(stream, fm1p, d2, nullptr, res2, B * V2, 128, 64, 64, 128, 128, 0, 0);
  run_conv(stream, idx2, v2b, fm1p, w2, b2, dir2, B, V2, KNN, 64, 128, fo, convA, meanb, istdb, fm2);
  run_sa(stream, fm2, B, V2, 128, sa[1][0], sa[1][1], sa[1][2], sa[1][3], sa[1][4], sa[1][5], sa[1][6],
         xq, xvt, xrt, diffb, tt, energy, cs, meanb, istdb);
  ew_add_relu_kernel<<<dim3(cdivu((size_t)B * V2 * 128, 256)), dim3(256), 0, stream>>>(fm2, fm2, res2, (size_t)B * V2 * 128);

  // ---- stage 3 (512 pts, 128->256, same graph) ----
  gemm(stream, fm2, d3, nullptr, res3, B * V2, 256, 128, 128, 256, 256, 0, 0);
  run_conv(stream, idx2, v2b, fm2, w3, b3, dir3, B, V2, KNN, 128, 256, fo, convA, meanb, istdb, fm3);
  run_sa(stream, fm3, B, V2, 256, sa[2][0], sa[2][1], sa[2][2], sa[2][3], sa[2][4], sa[2][5], sa[2][6],
         xq, xvt, xrt, diffb, tt, energy, cs, meanb, istdb);
  ew_add_relu_kernel<<<dim3(cdivu((size_t)B * V2 * 256, 256)), dim3(256), 0, stream>>>(fm3, fm3, res3, (size_t)B * V2 * 256);

  // ---- pool 2: 512 -> 128 ----
  knn_kernel<<<dim3(cdivu((size_t)B * V2, 128)), dim3(128), 0, stream>>>(v2b, idxp2, B, V2, 4);
  pool_kernel<<<dim3(B * V3), dim3(128), 0, stream>>>(fm3, v2b, idxp2, perm2, fm3p, v3b, B, V2, V3, 256);
  knn_kernel<<<dim3(cdivu((size_t)B * V3, 128)), dim3(128), 0, stream>>>(v3b, idx3, B, V3, KNN);

  // ---- stage 4 (128 pts, 256->1024) ----
  gemm(stream, fm3p, d4, nullptr, res4, B * V3, 1024, 256, 256, 1024, 1024, 0, 0);
  run_conv(stream, idx3, v3b, fm3p, w4, b4, dir4, B, V3, KNN, 256, 1024, fo, convA, meanb, istdb, fm4);
  run_sa(stream, fm4, B, V3, 1024, sa[3][0], sa[3][1], sa[3][2], sa[3][3], sa[3][4], sa[3][5], sa[3][6],
         xq, xvt, xrt, diffb, tt, energy, cs, meanb, istdb);
  ew_add_kernel<<<dim3(cdivu((size_t)B * V3 * 1024, 256)), dim3(256), 0, stream>>>(fm4, fm4, res4, (size_t)B * V3 * 1024);

  // ---- head ----
  maxover_points_kernel<<<dim3(cdivu((size_t)B * 1024, 256)), dim3(256), 0, stream>>>(fm4, gbuf, B, V3, 1024);
  gemm(stream, gbuf, cw1, cb1, hbuf, B, 256, 1024, 1024, 256, 256, 0, 0);
  colstats_kernel<<<dim3(256), dim3(256), 0, stream>>>(hbuf, meanb, istdb, B, 256, 1e-5f);
  norm_apply_kernel<<<dim3(cdivu((size_t)B * 256, 256)), dim3(256), 0, stream>>>(
      hbuf, meanb, istdb, cg, cbeta, nullptr, hbuf, (size_t)B * 256, 256, 1);
  gemm(stream, hbuf, cw2, cb2, outp, B, 40, 256, 256, 40, 40, 0, 0);
}
